// LinearTransformerLayer_59605556134023
// MI455X (gfx1250) — matrix operand order repaired and checked by reading, not compiled, not run on hardware
//
#include <hip/hip_runtime.h>
#include <hip/hip_bf16.h>

// ---------------------------------------------------------------------------
// CDNA5 (gfx1250) fused "linear transformer" layer.
// B=8, N=4096, C=512.  All GEMMs via v_wmma_f32_16x16x32_bf16 (wave32).
// MLP weight panels staged into LDS with the Tensor Data Mover (TDM),
// with TDM load-padding to kill LDS bank conflicts (row stride 1040 B).
// ---------------------------------------------------------------------------

typedef __attribute__((ext_vector_type(16))) __bf16 v16bf;
typedef __attribute__((ext_vector_type(8)))  float  v8f;
typedef __attribute__((ext_vector_type(4)))  unsigned int uint32x4;
typedef __attribute__((ext_vector_type(8)))  int          int32x8;
typedef __attribute__((ext_vector_type(4)))  int          int32x4;

#define Bb 8
#define Nn 4096
#define Cc 512
#define MT 32768        // B*N
#define LDSROW 520      // 512 bf16 + 8 pad (16 B) -> 260 DWORDs, bank-conflict-free

// ---- bf16 helpers (RNE, bit-level) -----------------------------------------
__device__ __forceinline__ unsigned short bfbits(float f) {
  unsigned int u = __builtin_bit_cast(unsigned int, f);
  unsigned int r = u + 0x7FFFu + ((u >> 16) & 1u);
  return (unsigned short)(r >> 16);
}

// ---- WMMA wrapper ----------------------------------------------------------
__device__ __forceinline__ v8f wmma_bf16(v16bf a, v16bf b, v8f c) {
  return __builtin_amdgcn_wmma_f32_16x16x32_bf16(
      false, a, false, b, (short)0, c, false, false);
}

// ---- fragment loaders ------------------------------------------------------
// A (16x32): lane l<16 -> row l, K = k0..k0+15;
//            lane l>=16 -> row l-16, K = k0+16..k0+31.
__device__ __forceinline__ v16bf ldA_bf(const unsigned short* A, int lda,
                                        int m0, int k0, int lrow, int lhi) {
  const unsigned short* p = A + (size_t)(m0 + lrow) * lda + k0 + lhi * 16;
  union { v16bf v; uint4 q[2]; } u;
  u.q[0] = *(const uint4*)p;
  u.q[1] = *(const uint4*)(p + 8);
  return u.v;
}

// B (32x16) from Bt stored [Ncols, Kdim] row-major:
// lane l<16 -> col l, K = k0..k0+15 ; lane l>=16 -> col l-16, K = k0+16..k0+31
__device__ __forceinline__ v16bf ldB_bf(const unsigned short* Bt, int ldb,
                                        int n0, int k0, int lrow, int lhi) {
  const unsigned short* p = Bt + (size_t)(n0 + lrow) * ldb + k0 + lhi * 16;
  union { v16bf v; uint4 q[2]; } u;
  u.q[0] = *(const uint4*)p;
  u.q[1] = *(const uint4*)(p + 8);
  return u.v;
}

// ---- generic inner loops ---------------------------------------------------
// 64x32 per wave (single-output GEMMs): 8 WMMA per 6 fragment loads
__device__ __forceinline__ void mm64x32_bf(const unsigned short* A, int lda,
                                           const unsigned short* Bt, int ldb,
                                           int m0, int n0, int kdim,
                                           int lrow, int lhi, v8f acc[4][2]) {
  for (int k0 = 0; k0 < kdim; k0 += 32) {
    if (k0 + 32 < kdim)
      __builtin_prefetch(A + (size_t)(m0 + lrow) * lda + k0 + 32, 0, 1);
    v16bf a[4];
    #pragma unroll
    for (int i = 0; i < 4; ++i) a[i] = ldA_bf(A, lda, m0 + 16 * i, k0, lrow, lhi);
    v16bf b0 = ldB_bf(Bt, ldb, n0,      k0, lrow, lhi);
    v16bf b1 = ldB_bf(Bt, ldb, n0 + 16, k0, lrow, lhi);
    #pragma unroll
    for (int i = 0; i < 4; ++i) {
      acc[i][0] = wmma_bf16(a[i], b0, acc[i][0]);
      acc[i][1] = wmma_bf16(a[i], b1, acc[i][1]);
    }
  }
}

// ---------------------------------------------------------------------------
// TDM: stage a 64-row x 512-col bf16 weight panel into LDS with row padding.
// D# per ISA ch.8: group0 {count=1, lds_addr, global_addr, type=2},
// group1 {data_size=2B, pad_enable, pad_interval=7 (256 DW), pad_amount=3
//         (4 DW = 16 B), tensor_dim0=512, tile 512x64, dim0_stride=512}.
// 6-arg builtin (clang-23 / therock-10.0).
// ---------------------------------------------------------------------------
__device__ __forceinline__ void tdm_load_panel(const unsigned short* gsrc,
                                               unsigned lds_off) {
  unsigned long long ga = (unsigned long long)(uintptr_t)gsrc;
  uint32x4 g0;
  g0[0] = 1u;                                    // count=1, user mode
  g0[1] = lds_off;                               // lds_addr (bytes)
  g0[2] = (unsigned)(ga & 0xFFFFFFFFu);          // global_addr[31:0]
  g0[3] = ((unsigned)(ga >> 32) & 0x01FFFFFFu) | 0x80000000u;  // addr[56:32] | type=2
  int32x8 g1;
  g1[0] = (int)((1u << 16) |       // data_size=1 (2 bytes); workgroup_mask=0
                (1u << 20) |       // pad_enable
                (7u << 22) |       // pad_interval: 256 DWORDs (one 1024B row)
                (3u << 25));       // pad_amount: 4 DWORDs (16 B)
  g1[1] = (int)(512u << 16);       // tensor_dim0[15:0]=512 (abar addr=0)
  g1[2] = (int)(0xFFFFu << 16);    // tensor_dim1[15:0]=0xFFFF (large; no OOB)
  g1[3] = (int)(512u << 16);       // tensor_dim1 hi=0, tile_dim0=512
  g1[4] = (int)64;                 // tile_dim1=64, tile_dim2=0
  g1[5] = (int)512;                // tensor_dim0_stride[31:0]=512
  g1[6] = 0;                       // stride hi / dim1_stride lo
  g1[7] = 0;
  int32x4 gz4 = {0, 0, 0, 0};
  int32x8 gz8 = {0, 0, 0, 0, 0, 0, 0, 0};
  __builtin_amdgcn_tensor_load_to_lds(g0, g1, gz4, gz4, gz8, 0);
}

// ---------------------------------------------------------------------------
// Kernel 0: fp32 -> bf16 conversion, 8 elements/thread (vectorized)
// ---------------------------------------------------------------------------
__global__ __launch_bounds__(256) void k_cvt8(const float* __restrict__ src,
                                              unsigned short* __restrict__ dst, int n8) {
  int i = blockIdx.x * 256 + threadIdx.x;
  if (i >= n8) return;
  const float4* s = (const float4*)(src + (size_t)i * 8);
  float4 a = s[0], b = s[1];
  union { unsigned short h[8]; uint4 q; } u;
  u.h[0] = bfbits(a.x); u.h[1] = bfbits(a.y); u.h[2] = bfbits(a.z); u.h[3] = bfbits(a.w);
  u.h[4] = bfbits(b.x); u.h[5] = bfbits(b.y); u.h[6] = bfbits(b.z); u.h[7] = bfbits(b.w);
  *(uint4*)(dst + (size_t)i * 8) = u.q;
}

// ---------------------------------------------------------------------------
// Kernel 1: fused QKV projection.  Block tile 128x64 (8 waves of 32x32, 3 outs)
//   qlog  : fp32 [MT, C]       ktlog : fp32 [B, C, N]      vtb : bf16 [B, C, N]
// ---------------------------------------------------------------------------
__global__ __launch_bounds__(256) void k_qkv(const unsigned short* __restrict__ xb,
                                             const unsigned short* __restrict__ wq,
                                             const unsigned short* __restrict__ wk,
                                             const unsigned short* __restrict__ wv,
                                             float* __restrict__ qlog,
                                             float* __restrict__ ktlog,
                                             unsigned short* __restrict__ vtb) {
  int lane = threadIdx.x & 31, w = threadIdx.x >> 5;
  int wm = w & 3, wn = w >> 2;
  int m0 = blockIdx.x * 128 + wm * 32;
  int n0 = blockIdx.y * 64 + wn * 32;
  int lrow = lane & 15, lhi = lane >> 4;

  v8f aq[2][2] = {}, ak[2][2] = {}, av[2][2] = {};
  for (int k0 = 0; k0 < Cc; k0 += 32) {
    v16bf a0 = ldA_bf(xb, Cc, m0,      k0, lrow, lhi);
    v16bf a1 = ldA_bf(xb, Cc, m0 + 16, k0, lrow, lhi);
    v16bf q0 = ldB_bf(wq, Cc, n0,      k0, lrow, lhi);
    v16bf q1 = ldB_bf(wq, Cc, n0 + 16, k0, lrow, lhi);
    v16bf kk0 = ldB_bf(wk, Cc, n0,      k0, lrow, lhi);
    v16bf kk1 = ldB_bf(wk, Cc, n0 + 16, k0, lrow, lhi);
    v16bf v0 = ldB_bf(wv, Cc, n0,      k0, lrow, lhi);
    v16bf v1 = ldB_bf(wv, Cc, n0 + 16, k0, lrow, lhi);
    aq[0][0] = wmma_bf16(a0, q0, aq[0][0]);  aq[0][1] = wmma_bf16(a0, q1, aq[0][1]);
    aq[1][0] = wmma_bf16(a1, q0, aq[1][0]);  aq[1][1] = wmma_bf16(a1, q1, aq[1][1]);
    ak[0][0] = wmma_bf16(a0, kk0, ak[0][0]); ak[0][1] = wmma_bf16(a0, kk1, ak[0][1]);
    ak[1][0] = wmma_bf16(a1, kk0, ak[1][0]); ak[1][1] = wmma_bf16(a1, kk1, ak[1][1]);
    av[0][0] = wmma_bf16(a0, v0, av[0][0]);  av[0][1] = wmma_bf16(a0, v1, av[0][1]);
    av[1][0] = wmma_bf16(a1, v0, av[1][0]);  av[1][1] = wmma_bf16(a1, v1, av[1][1]);
  }

  #pragma unroll
  for (int ti = 0; ti < 2; ++ti)
    #pragma unroll
    for (int tj = 0; tj < 2; ++tj)
      #pragma unroll
      for (int v = 0; v < 8; ++v) {
        int m = m0 + 16 * ti + 8 * lhi + v;
        int c = n0 + 16 * tj + lrow;
        qlog[(size_t)m * Cc + c] = aq[ti][tj][v];
        int b = m >> 12, ns = m & (Nn - 1);
        size_t tpos = ((size_t)(b * Cc + c) << 12) + ns;
        ktlog[tpos] = ak[ti][tj][v];
        vtb[tpos]   = bfbits(av[ti][tj][v]);
      }
}

// ---------------------------------------------------------------------------
// Kernel 2: channel softmax for Q (rows of 512) * C^-0.5 -> bf16; wave/row
// ---------------------------------------------------------------------------
__global__ __launch_bounds__(256) void k_qsoftmax(const float* __restrict__ qlog,
                                                  unsigned short* __restrict__ qb) {
  const float scale = 0.04419417382415922f;  // 512^-0.5
  int lane = threadIdx.x & 31;
  int row = blockIdx.x * 8 + (threadIdx.x >> 5);
  const float* src = qlog + (size_t)row * Cc;
  float v[16];
  float mx = -3.4e38f;
  #pragma unroll
  for (int i = 0; i < 16; ++i) { v[i] = src[lane + 32 * i]; mx = fmaxf(mx, v[i]); }
  #pragma unroll
  for (int off = 16; off > 0; off >>= 1) mx = fmaxf(mx, __shfl_xor(mx, off, 32));
  float s = 0.f;
  #pragma unroll
  for (int i = 0; i < 16; ++i) { v[i] = expf(v[i] - mx); s += v[i]; }
  #pragma unroll
  for (int off = 16; off > 0; off >>= 1) s += __shfl_xor(s, off, 32);
  float inv = scale / s;
  unsigned short* dst = qb + (size_t)row * Cc;
  #pragma unroll
  for (int i = 0; i < 16; ++i) dst[lane + 32 * i] = bfbits(v[i] * inv);
}

// ---------------------------------------------------------------------------
// Kernel 3: sequence softmax for K over N=4096 -> bf16; block/row, LDS reduce
// ---------------------------------------------------------------------------
__global__ __launch_bounds__(256) void k_ksoftmax(const float* __restrict__ ktlog,
                                                  unsigned short* __restrict__ ktb) {
  __shared__ float red[256];
  int tid = threadIdx.x;
  const float* src = ktlog + (size_t)blockIdx.x * Nn;
  float v[16];
  float mx = -3.4e38f;
  #pragma unroll
  for (int i = 0; i < 16; ++i) { v[i] = src[tid + 256 * i]; mx = fmaxf(mx, v[i]); }
  red[tid] = mx; __syncthreads();
  for (int s = 128; s > 0; s >>= 1) {
    if (tid < s) red[tid] = fmaxf(red[tid], red[tid + s]);
    __syncthreads();
  }
  mx = red[0]; __syncthreads();
  float sum = 0.f;
  #pragma unroll
  for (int i = 0; i < 16; ++i) { v[i] = expf(v[i] - mx); sum += v[i]; }
  red[tid] = sum; __syncthreads();
  for (int s = 128; s > 0; s >>= 1) {
    if (tid < s) red[tid] += red[tid + s];
    __syncthreads();
  }
  float inv = 1.0f / red[0];
  unsigned short* dst = ktb + (size_t)blockIdx.x * Nn;
  #pragma unroll
  for (int i = 0; i < 16; ++i) dst[tid + 256 * i] = bfbits(v[i] * inv);
}

// ---------------------------------------------------------------------------
// Kernel 4: context[b] = K^T V ; block 128x128 (8 waves of 64x32), out ctxT bf16
// ---------------------------------------------------------------------------
__global__ __launch_bounds__(256) void k_context(const unsigned short* __restrict__ ktb,
                                                 const unsigned short* __restrict__ vtb,
                                                 unsigned short* __restrict__ ctxT) {
  int lane = threadIdx.x & 31, w = threadIdx.x >> 5;
  int wm = w & 1, wn = w >> 1;
  int b = blockIdx.z;
  int m0 = blockIdx.x * 128 + wm * 64;
  int n0 = blockIdx.y * 128 + wn * 32;
  int lrow = lane & 15, lhi = lane >> 4;
  const unsigned short* A  = ktb + (size_t)b * Cc * Nn;
  const unsigned short* Bt = vtb + (size_t)b * Cc * Nn;
  v8f acc[4][2] = {};
  mm64x32_bf(A, Nn, Bt, Nn, m0, n0, Nn, lrow, lhi, acc);
  #pragma unroll
  for (int ti = 0; ti < 4; ++ti)
    #pragma unroll
    for (int tj = 0; tj < 2; ++tj)
      #pragma unroll
      for (int v = 0; v < 8; ++v) {
        int d = m0 + 16 * ti + 8 * lhi + v;
        int e = n0 + 16 * tj + lrow;
        ctxT[((size_t)b * Cc + e) * Cc + d] = bfbits(acc[ti][tj][v]);
      }
}

// ---------------------------------------------------------------------------
// Kernel 5: attn = Q @ context ; y = attn + x ; block 128x128
// ---------------------------------------------------------------------------
__global__ __launch_bounds__(256) void k_attn(const unsigned short* __restrict__ qb,
                                              const unsigned short* __restrict__ ctxT,
                                              const float* __restrict__ x,
                                              float* __restrict__ y,
                                              unsigned short* __restrict__ yb) {
  int lane = threadIdx.x & 31, w = threadIdx.x >> 5;
  int wm = w & 1, wn = w >> 1;
  int m0 = blockIdx.x * 128 + wm * 64;
  int n0 = blockIdx.y * 128 + wn * 32;
  int lrow = lane & 15, lhi = lane >> 4;
  int b = blockIdx.x >> 5;  // 32 blocks of 128 rows per batch
  const unsigned short* Bt = ctxT + (size_t)b * Cc * Cc;
  v8f acc[4][2] = {};
  mm64x32_bf(qb, Cc, Bt, Cc, m0, n0, Cc, lrow, lhi, acc);
  #pragma unroll
  for (int ti = 0; ti < 4; ++ti)
    #pragma unroll
    for (int tj = 0; tj < 2; ++tj)
      #pragma unroll
      for (int v = 0; v < 8; ++v) {
        int m = m0 + 16 * ti + 8 * lhi + v;
        int c = n0 + 16 * tj + lrow;
        size_t pos = (size_t)m * Cc + c;
        float val = acc[ti][tj][v] + x[pos];
        y[pos]  = val;
        yb[pos] = bfbits(val);
      }
}

// ---------------------------------------------------------------------------
// Kernels 6/7: MLP GEMMs with TDM-staged weight panels (padded rows in LDS).
// Block tile 256x64 (8 waves: wm 0..3 x 64 rows, wn 0..1 x 32 cols).
// ---------------------------------------------------------------------------
__device__ __forceinline__ v16bf ldB_lds(const unsigned short* bsm,
                                         int n0, int k0, int lrow, int lhi) {
  const unsigned short* p = bsm + (size_t)(n0 + lrow) * LDSROW + k0 + lhi * 16;
  union { v16bf v; uint4 q[2]; } u;
  u.q[0] = *(const uint4*)p;
  u.q[1] = *(const uint4*)(p + 8);
  return u.v;
}

__device__ __forceinline__ void mlp_gemm(const unsigned short* A,
                                         const unsigned short* wpanel_lds,
                                         int m0, int n0loc,
                                         int lrow, int lhi, v8f acc[4][2]) {
  for (int k0 = 0; k0 < Cc; k0 += 32) {
    if (k0 + 32 < Cc)
      __builtin_prefetch(A + (size_t)(m0 + lrow) * Cc + k0 + 32, 0, 1);
    v16bf a[4];
    #pragma unroll
    for (int i = 0; i < 4; ++i) a[i] = ldA_bf(A, Cc, m0 + 16 * i, k0, lrow, lhi);
    v16bf b0 = ldB_lds(wpanel_lds, n0loc,      k0, lrow, lhi);
    v16bf b1 = ldB_lds(wpanel_lds, n0loc + 16, k0, lrow, lhi);
    #pragma unroll
    for (int i = 0; i < 4; ++i) {
      acc[i][0] = wmma_bf16(a[i], b0, acc[i][0]);
      acc[i][1] = wmma_bf16(a[i], b1, acc[i][1]);
    }
  }
}

__global__ __launch_bounds__(256) void k_mlp1(const unsigned short* __restrict__ yb,
                                              const unsigned short* __restrict__ w1,
                                              unsigned short* __restrict__ hb) {
  __shared__ unsigned short bsm[64 * LDSROW];
  int lane = threadIdx.x & 31, w = threadIdx.x >> 5;
  int wm = w >> 1, wn = w & 1;
  int m0 = blockIdx.x * 256 + wm * 64;
  int n0loc = wn * 32;
  int n0 = blockIdx.y * 64 + n0loc;
  int lrow = lane & 15, lhi = lane >> 4;

  if (threadIdx.x < 32) {
    tdm_load_panel(w1 + (size_t)blockIdx.y * 64 * Cc, (unsigned)(uintptr_t)&bsm[0]);
    __builtin_amdgcn_s_wait_tensorcnt(0);
  }
  __syncthreads();

  v8f acc[4][2] = {};
  mlp_gemm(yb, bsm, m0, n0loc, lrow, lhi, acc);
  #pragma unroll
  for (int ti = 0; ti < 4; ++ti)
    #pragma unroll
    for (int tj = 0; tj < 2; ++tj)
      #pragma unroll
      for (int v = 0; v < 8; ++v) {
        int m = m0 + 16 * ti + 8 * lhi + v;
        int c = n0 + 16 * tj + lrow;
        hb[(size_t)m * Cc + c] = bfbits(acc[ti][tj][v]);
      }
}

__global__ __launch_bounds__(256) void k_mlp2(const unsigned short* __restrict__ hb,
                                              const unsigned short* __restrict__ w2,
                                              const float* __restrict__ y,
                                              float* __restrict__ out) {
  __shared__ unsigned short bsm[64 * LDSROW];
  int lane = threadIdx.x & 31, w = threadIdx.x >> 5;
  int wm = w >> 1, wn = w & 1;
  int m0 = blockIdx.x * 256 + wm * 64;
  int n0loc = wn * 32;
  int n0 = blockIdx.y * 64 + n0loc;
  int lrow = lane & 15, lhi = lane >> 4;

  if (threadIdx.x < 32) {
    tdm_load_panel(w2 + (size_t)blockIdx.y * 64 * Cc, (unsigned)(uintptr_t)&bsm[0]);
    __builtin_amdgcn_s_wait_tensorcnt(0);
  }
  __syncthreads();

  v8f acc[4][2] = {};
  mlp_gemm(hb, bsm, m0, n0loc, lrow, lhi, acc);
  #pragma unroll
  for (int ti = 0; ti < 4; ++ti)
    #pragma unroll
    for (int tj = 0; tj < 2; ++tj)
      #pragma unroll
      for (int v = 0; v < 8; ++v) {
        int m = m0 + 16 * ti + 8 * lhi + v;
        int c = n0 + 16 * tj + lrow;
        size_t pos = (size_t)m * Cc + c;
        out[pos] = acc[ti][tj][v] + y[pos];
      }
}

// ---------------------------------------------------------------------------
// Host-side launcher
// ---------------------------------------------------------------------------
extern "C" void kernel_launch(void* const* d_in, const int* in_sizes, int n_in,
                              void* d_out, int out_size, void* d_ws, size_t ws_size,
                              hipStream_t stream) {
  const float* x  = (const float*)d_in[0];
  const float* Wq = (const float*)d_in[1];
  const float* Wk = (const float*)d_in[2];
  const float* Wv = (const float*)d_in[3];
  const float* W1 = (const float*)d_in[4];
  const float* W2 = (const float*)d_in[5];
  float* out = (float*)d_out;

  char* ws = (char*)d_ws;
  const size_t F32T  = (size_t)MT * Cc * 4;  // 64 MiB fp32 tensor
  const size_t BF16T = (size_t)MT * Cc * 2;  // 32 MiB bf16 tensor
  size_t off = 0;
  float*          qlog  = (float*)(ws + off);          off += F32T;
  float*          ktlog = (float*)(ws + off);          off += F32T;
  float*          y     = (float*)(ws + off);          off += F32T;
  unsigned short* xb    = (unsigned short*)(ws + off); off += BF16T;
  unsigned short* qb    = (unsigned short*)(ws + off); off += BF16T;
  unsigned short* ktb   = (unsigned short*)(ws + off); off += BF16T;
  unsigned short* vtb   = (unsigned short*)(ws + off); off += BF16T;
  unsigned short* yb    = (unsigned short*)(ws + off); off += BF16T;
  unsigned short* hb    = (unsigned short*)(ws + off); off += BF16T;
  unsigned short* ctxT  = (unsigned short*)(ws + off); off += (size_t)Bb * Cc * Cc * 2;
  unsigned short* wqb   = (unsigned short*)(ws + off); off += (size_t)Cc * Cc * 2;
  unsigned short* wkb   = (unsigned short*)(ws + off); off += (size_t)Cc * Cc * 2;
  unsigned short* wvb   = (unsigned short*)(ws + off); off += (size_t)Cc * Cc * 2;
  unsigned short* w1b   = (unsigned short*)(ws + off); off += (size_t)Cc * Cc * 2;
  unsigned short* w2b   = (unsigned short*)(ws + off); off += (size_t)Cc * Cc * 2;

  const int WN8 = Cc * Cc / 8;   // weight elements / 8
  const int XN8 = MT * Cc / 8;   // x elements / 8
  k_cvt8<<<(WN8 + 255) / 256, 256, 0, stream>>>(Wq, wqb, WN8);
  k_cvt8<<<(WN8 + 255) / 256, 256, 0, stream>>>(Wk, wkb, WN8);
  k_cvt8<<<(WN8 + 255) / 256, 256, 0, stream>>>(Wv, wvb, WN8);
  k_cvt8<<<(WN8 + 255) / 256, 256, 0, stream>>>(W1, w1b, WN8);
  k_cvt8<<<(WN8 + 255) / 256, 256, 0, stream>>>(W2, w2b, WN8);
  k_cvt8<<<(XN8 + 255) / 256, 256, 0, stream>>>(x, xb, XN8);

  k_qkv<<<dim3(MT / 128, Cc / 64), 256, 0, stream>>>(xb, wqb, wkb, wvb, qlog, ktlog, vtb);
  k_qsoftmax<<<MT / 8, 256, 0, stream>>>(qlog, qb);
  k_ksoftmax<<<Bb * Cc, 256, 0, stream>>>(ktlog, ktb);
  k_context<<<dim3(Cc / 128, Cc / 128, Bb), 256, 0, stream>>>(ktb, vtb, ctxT);
  k_attn<<<dim3(MT / 128, Cc / 128), 256, 0, stream>>>(qb, ctxT, x, y, yb);
  k_mlp1<<<dim3(MT / 256, Cc / 64), 256, 0, stream>>>(yb, w1b, hb);
  k_mlp2<<<dim3(MT / 256, Cc / 64), 256, 0, stream>>>(hb, w2b, y, out);
}
